// Block_5420248727653
// MI455X (gfx1250) — compile-verified, matrix operand order checked by reading, not run on hardware
//
#include <hip/hip_runtime.h>
#include <hip/hip_bf16.h>
#include <math.h>

typedef __bf16 bf16;
typedef __attribute__((ext_vector_type(16))) __bf16 v16bf;
typedef __attribute__((ext_vector_type(8)))  float  v8f;
typedef unsigned int u32x4 __attribute__((ext_vector_type(4)));
typedef int          i32x8 __attribute__((ext_vector_type(8)));

constexpr int kNS = 50000;
constexpr int kNT = 50000;
constexpr int kE  = 600000;
constexpr int kD  = 128;
constexpr float kEPS = 1.1920928955078125e-7f; // finfo(f32).eps

// ---------------------------------------------------------------- WMMA utils
// 16x16x32 bf16 fragment load from a row-major [row][k] tile (LDS or global).
// Lane 0-15: row=lane,  K = kb..kb+7 and kb+16..kb+23 with kb=0
// Lane16-31: row=lane-16, same pattern with kb=8  (per CDNA5 ISA 7.12.2)
__device__ __forceinline__ v16bf ld_frag(const bf16* __restrict__ p, int stride, int lane) {
  int r  = lane & 15;
  int kb = (lane >> 4) << 3;
  const bf16* q = p + (size_t)r * stride + kb;
  v16bf f;
#pragma unroll
  for (int i = 0; i < 8; ++i) f[i] = q[i];
#pragma unroll
  for (int i = 0; i < 8; ++i) f[8 + i] = q[16 + i];
  return f;
}

__device__ __forceinline__ v8f wmma_bf16(v16bf a, v16bf b, v8f c) {
  return __builtin_amdgcn_wmma_f32_16x16x32_bf16(false, a, false, b, (short)0, c, false, false);
}

__device__ __forceinline__ float leaky(float x, float s) { return x >= 0.f ? x : s * x; }

__device__ __forceinline__ void atomicMaxF(float* addr, float v) {
  int* ai = (int*)addr;
  int cur = __float_as_int(*addr);
  while (__int_as_float(cur) < v) {
    int prev = atomicCAS(ai, cur, __float_as_int(v));
    if (prev == cur) break;
    cur = prev;
  }
}

// ---------------------------------------------------------------- TDM helper
// Issue a TENSOR_LOAD_TO_LDS of a [min(64,remRows)] x [64] bf16 tile from a
// row-major matrix (row stride = strideElems). LDS padding fields add 16 bytes
// after every 128-byte row -> LDS tile stride 72 bf16. D# per CDNA5 ISA ch.8.
__device__ __forceinline__ void tdm_load_tile(const bf16* gsrc, unsigned ldsAddr,
                                              int remRows, int remCols, int strideElems) {
  unsigned long long ga = (unsigned long long)(uintptr_t)gsrc;
  u32x4 g0;
  g0[0] = 1u;                                                  // count=1, user descriptor
  g0[1] = ldsAddr;                                             // LDS byte address
  g0[2] = (unsigned)ga;                                        // global_addr[31:0]
  g0[3] = (unsigned)((ga >> 32) & 0x01FFFFFFu) | (2u << 30);   // global_addr[56:32] | type=2
  i32x8 g1;
  // data_size=2B, pad_enable, pad_interval=32 DWORDs (code 4), pad_amount=4 DWORDs (code 3)
  g1[0] = (int)((1u << 16) | (1u << 20) | (4u << 22) | (3u << 25));
  g1[1] = (int)(((unsigned)remCols & 0xFFFFu) << 16);          // tensor_dim0[15:0]
  g1[2] = (int)((((unsigned)remCols >> 16) & 0xFFFFu) |        // tensor_dim0[31:16]
                (((unsigned)remRows & 0xFFFFu) << 16));        // tensor_dim1[15:0]
  g1[3] = (int)((((unsigned)remRows >> 16) & 0xFFFFu) |        // tensor_dim1[31:16]
                (64u << 16));                                  // tile_dim0 = 64
  g1[4] = 64;                                                  // tile_dim1 = 64, tile_dim2 = 0
  g1[5] = strideElems;                                         // tensor_dim0_stride[31:0]
  g1[6] = 0;
  g1[7] = 0;
  asm volatile("tensor_load_to_lds %0, %1" :: "s"(g0), "s"(g1) : "memory");
}

// ---------------------------------------------------------------- tiny utils
__global__ void f2bf_kernel(const float* __restrict__ in, bf16* __restrict__ out, int n) {
  int i = blockIdx.x * 256 + threadIdx.x;
  if (i < n) out[i] = (bf16)in[i];
}

__global__ void wcat_kernel(const float* __restrict__ Ws, const float* __restrict__ Wt,
                            const float* __restrict__ We, bf16* __restrict__ out) {
  int i = blockIdx.x * 256 + threadIdx.x;
  if (i >= 128 * 384) return;
  int n = i / 384, k = i % 384;
  float v = (k < 128) ? Ws[n * 128 + k] : (k < 256) ? Wt[n * 128 + k - 128] : We[n * 128 + k - 256];
  out[i] = (bf16)v;
}

__global__ void i2f_kernel(const int* __restrict__ in, float* __restrict__ out, int n) {
  int i = blockIdx.x * 256 + threadIdx.x;
  if (i < n) out[i] = (float)in[i];
}

__global__ void fill_kernel(float* __restrict__ p, int n, float v) {
  int i = blockIdx.x * 256 + threadIdx.x;
  if (i < n) p[i] = v;
}

// ---------------------------------------------------------------- edge h GEMM
// h[e,:] = [xs[src], xt[tgt], ea[e]] (384) @ Wcat^T (128x384), tile 64 edges.
__global__ __launch_bounds__(256) void edge_h_kernel(
    const float* __restrict__ xs, const float* __restrict__ xt, const int* __restrict__ ei,
    const float* __restrict__ ea, const bf16* __restrict__ Wcat, float* __restrict__ Hout) {
  __shared__ bf16 Asl[64 * 72];
  __shared__ int sidx[64], tidx[64];
  int eBase = blockIdx.x * 64;
  if (threadIdx.x < 64) {
    int e = eBase + threadIdx.x;
    sidx[threadIdx.x] = ei[e];
    tidx[threadIdx.x] = ei[kE + e];
  }
  __syncthreads();
  int wave = threadIdx.x >> 5, lane = threadIdx.x & 31;
  int rowTile = wave & 3;
  int colBase = (wave >> 2) * 64;
  v8f acc[4] = {};
  for (int ks = 0; ks < 384; ks += 64) {
    int r = threadIdx.x >> 2;
    int c0 = (threadIdx.x & 3) * 16;
    int k = ks + c0;
    const float* srcp;
    if (k < 128)      srcp = &xs[(size_t)sidx[r] * kD + k];
    else if (k < 256) srcp = &xt[(size_t)tidx[r] * kD + (k - 128)];
    else              srcp = &ea[(size_t)(eBase + r) * kD + (k - 256)];
    bf16* dst = &Asl[r * 72 + c0];
#pragma unroll
    for (int i = 0; i < 16; ++i) dst[i] = (bf16)srcp[i];
    __syncthreads();
#pragma unroll
    for (int kk = 0; kk < 64; kk += 32) {
      v16bf a = ld_frag(&Asl[rowTile * 16 * 72 + kk], 72, lane);
#pragma unroll
      for (int t = 0; t < 4; ++t) {
        v16bf b = ld_frag(&Wcat[(size_t)(colBase + t * 16) * 384 + ks + kk], 384, lane);
        acc[t] = wmma_bf16(a, b, acc[t]);
      }
    }
    __syncthreads();
  }
  int cr = 8 * (lane >> 4), cc = lane & 15;
#pragma unroll
  for (int t = 0; t < 4; ++t)
#pragma unroll
    for (int v = 0; v < 8; ++v) {
      int row = eBase + rowTile * 16 + cr + v;
      Hout[(size_t)row * kD + colBase + t * 16 + cc] = acc[t][v];
    }
}

// ---------------------------------------------------------------- msg MLP + scatter
// mode 0 (SourceModel): A0=[x_t[tgt], ea], scatter raw moments 1..4 keyed by src.
// mode 1 (TargetModel): A0=[x_s[src], ea], scatter sum keyed by tgt.
__global__ __launch_bounds__(256) void msg_kernel(
    const float* __restrict__ X, const float* __restrict__ ea, const int* __restrict__ ei,
    const bf16* __restrict__ W1, const float* __restrict__ b1,
    const bf16* __restrict__ W2, const float* __restrict__ b2,
    float* __restrict__ out0, int mode) {
  __shared__ bf16 Asl[64 * 72];
  __shared__ bf16 Mid[64 * 264];
  __shared__ int gidx[64], kidx[64];
  int eBase = blockIdx.x * 64;
  if (threadIdx.x < 64) {
    int e = eBase + threadIdx.x;
    int s = ei[e], t = ei[kE + e];
    gidx[threadIdx.x] = (mode == 0) ? t : s;
    kidx[threadIdx.x] = (mode == 0) ? s : t;
  }
  __syncthreads();
  int wave = threadIdx.x >> 5, lane = threadIdx.x & 31;
  int rowTile = wave & 3;
  int colBase = (wave >> 2) * 128;
  v8f acc[8] = {};
  // phase 1: mid = leaky0.2(A0 @ W1^T + b1), K=256
  for (int ks = 0; ks < 256; ks += 64) {
    int r = threadIdx.x >> 2;
    int c0 = (threadIdx.x & 3) * 16;
    int k = ks + c0;
    const float* srcp = (k < 128) ? &X[(size_t)gidx[r] * kD + k]
                                  : &ea[(size_t)(eBase + r) * kD + (k - 128)];
    bf16* dst = &Asl[r * 72 + c0];
#pragma unroll
    for (int i = 0; i < 16; ++i) dst[i] = (bf16)srcp[i];
    __syncthreads();
#pragma unroll
    for (int kk = 0; kk < 64; kk += 32) {
      v16bf a = ld_frag(&Asl[rowTile * 16 * 72 + kk], 72, lane);
#pragma unroll
      for (int t = 0; t < 8; ++t) {
        v16bf b = ld_frag(&W1[(size_t)(colBase + t * 16) * 256 + ks + kk], 256, lane);
        acc[t] = wmma_bf16(a, b, acc[t]);
      }
    }
    __syncthreads();
  }
  {
    int cr = 8 * (lane >> 4), cc = lane & 15;
#pragma unroll
    for (int t = 0; t < 8; ++t) {
      int col = colBase + t * 16 + cc;
      float bv = b1[col];
#pragma unroll
      for (int v = 0; v < 8; ++v) {
        float x = leaky(acc[t][v] + bv, 0.2f);
        Mid[(rowTile * 16 + cr + v) * 264 + col] = (bf16)x;
        acc[t][v] = 0.f;
      }
    }
  }
  __syncthreads();
  // phase 2: msg = Mid @ W2^T + b2, K=256
  for (int ks = 0; ks < 256; ks += 32) {
    v16bf a = ld_frag(&Mid[rowTile * 16 * 264 + ks], 264, lane);
#pragma unroll
    for (int t = 0; t < 8; ++t) {
      v16bf b = ld_frag(&W2[(size_t)(colBase + t * 16) * 256 + ks], 256, lane);
      acc[t] = wmma_bf16(a, b, acc[t]);
    }
  }
  {
    int cr = 8 * (lane >> 4), cc = lane & 15;
    const size_t seg = (size_t)kNS * 256;
#pragma unroll
    for (int t = 0; t < 8; ++t) {
      int col = colBase + t * 16 + cc;
      float bv = b2[col];
#pragma unroll
      for (int v = 0; v < 8; ++v) {
        int key = kidx[rowTile * 16 + cr + v];
        float x = acc[t][v] + bv;
        if (mode == 0) {
          float* m = out0 + (size_t)key * 256 + col;
          float x2 = x * x;
          atomicAdd(m, x);
          atomicAdd(m + seg, x2);
          atomicAdd(m + 2 * seg, x2 * x);
          atomicAdd(m + 3 * seg, x2 * x2);
        } else {
          atomicAdd(out0 + (size_t)key * 256 + col, x);
        }
      }
    }
  }
}

// ---------------------------------------------------------------- generic GEMM
// Out[M,N] = act(A[M,K](bf16) @ B[N,K]^T(bf16) + bias), N%128==0, K%64==0.
// A tiles are DMA'd into double-buffered LDS by the Tensor Data Mover; the
// DMA of tile k+1 overlaps the WMMA work on tile k (TENSORcnt sync).
template <typename OutT>
__global__ __launch_bounds__(256) void gemm_kernel(
    const bf16* __restrict__ A, const bf16* __restrict__ B, const float* __restrict__ bias,
    OutT* __restrict__ Out, int M, int N, int K, int act) {
  __shared__ bf16 Asl[2][64 * 72];
  int wave = threadIdx.x >> 5, lane = threadIdx.x & 31;
  int rowTile = wave & 3;
  int rowBase = blockIdx.x * 64;
  int colBase = blockIdx.y * 128 + (wave >> 2) * 64;
  v8f acc[4] = {};
  const int nk = K >> 6;
  unsigned lds0 = (unsigned)(uintptr_t)(void*)&Asl[0][0];
  unsigned lds1 = (unsigned)(uintptr_t)(void*)&Asl[1][0];
  const bf16* Arow = A + (size_t)rowBase * K;
  if (wave == 0) tdm_load_tile(Arow, lds0, M - rowBase, K, K);
  for (int kt = 0; kt < nk; ++kt) {
    const bf16* cur = &Asl[kt & 1][0];
    if (wave == 0) {
      if (kt + 1 < nk) {
        tdm_load_tile(Arow + (kt + 1) * 64, (kt & 1) ? lds0 : lds1,
                      M - rowBase, K - (kt + 1) * 64, K);
        __builtin_amdgcn_s_wait_tensorcnt(1);  // tile kt done; kt+1 in flight
      } else {
        __builtin_amdgcn_s_wait_tensorcnt(0);
      }
    }
    __syncthreads();
    int ks = kt * 64;
#pragma unroll
    for (int kk = 0; kk < 64; kk += 32) {
      v16bf a = ld_frag(&cur[rowTile * 16 * 72 + kk], 72, lane);
#pragma unroll
      for (int t = 0; t < 4; ++t) {
        v16bf b = ld_frag(&B[(size_t)(colBase + t * 16) * K + ks + kk], K, lane);
        acc[t] = wmma_bf16(a, b, acc[t]);
      }
    }
    __syncthreads();
  }
  int cr = 8 * (lane >> 4), cc = lane & 15;
#pragma unroll
  for (int t = 0; t < 4; ++t) {
    int col = colBase + t * 16 + cc;
    float bv = bias ? bias[col] : 0.f;
#pragma unroll
    for (int v = 0; v < 8; ++v) {
      int row = rowBase + rowTile * 16 + cr + v;
      if (row < M) {
        float x = acc[t][v] + bv;
        if (act) x = leaky(x, 0.2f);
        Out[(size_t)row * N + col] = (OutT)x;
      }
    }
  }
}

// ---------------------------------------------------------------- softmax passes
__global__ void escore_kernel(const float* __restrict__ h, const float* __restrict__ wattn,
                              float* __restrict__ esc) {
  int e = blockIdx.x * 8 + (threadIdx.x >> 5);
  int lane = threadIdx.x & 31;
  if (e >= kE) return;
  float s = 0.f;
#pragma unroll
  for (int i = 0; i < 4; ++i) {
    int c = lane + 32 * i;
    s += h[(size_t)e * kD + c] * wattn[c];
  }
  for (int off = 16; off; off >>= 1) s += __shfl_xor(s, off, 32);
  if (lane == 0) esc[e] = leaky(s, 0.2f);
}

__global__ void segmax_kernel(const int* __restrict__ ei, const float* __restrict__ esc,
                              float* __restrict__ smax) {
  int e = blockIdx.x * 256 + threadIdx.x;
  if (e < kE) atomicMaxF(&smax[ei[e]], esc[e]);
}

__global__ void expsum_kernel(const int* __restrict__ ei, float* __restrict__ esc,
                              const float* __restrict__ smax, float* __restrict__ ssum) {
  int e = blockIdx.x * 256 + threadIdx.x;
  if (e >= kE) return;
  float ex = expf(esc[e] - smax[ei[e]]);
  esc[e] = ex;
  atomicAdd(&ssum[ei[e]], ex);
}

__global__ void alpha_kernel(const int* __restrict__ ei, float* __restrict__ esc,
                             const float* __restrict__ ssum) {
  int e = blockIdx.x * 256 + threadIdx.x;
  if (e < kE) esc[e] = esc[e] / ssum[ei[e]];
}

__global__ void cnt_kernel(const int* __restrict__ ei, float* __restrict__ cnt) {
  int e = blockIdx.x * 256 + threadIdx.x;
  if (e < kE) atomicAdd(&cnt[ei[e]], 1.f);
}

// ---------------------------------------------------------------- RMS residual
// out[row,:] = base[row,:] + rmsnorm(x[row,:]*scale[row]) * g   (dim 128)
__global__ void rmsres_kernel(const float* __restrict__ base, const float* __restrict__ x,
                              const float* __restrict__ scale, const float* __restrict__ g,
                              float* __restrict__ out, int rows) {
  int row = blockIdx.x * 8 + (threadIdx.x >> 5);
  int lane = threadIdx.x & 31;
  if (row >= rows) return;
  float sc = scale ? scale[row] : 1.f;
  float v[4];
  float ss = 0.f;
#pragma unroll
  for (int i = 0; i < 4; ++i) {
    v[i] = x[(size_t)row * kD + lane + 32 * i] * sc;
    ss += v[i] * v[i];
  }
  for (int off = 16; off; off >>= 1) ss += __shfl_xor(ss, off, 32);
  float rr = rsqrtf(ss * (1.f / 128.f) + kEPS);
#pragma unroll
  for (int i = 0; i < 4; ++i) {
    int c = lane + 32 * i;
    out[(size_t)row * kD + c] = base[(size_t)row * kD + c] + v[i] * rr * g[c];
  }
}

// ---------------------------------------------------------------- feature assembly
__global__ void hs_build_kernel(const float* __restrict__ xs, const float* __restrict__ mom,
                                const float* __restrict__ cnt, const float* __restrict__ xu,
                                bf16* __restrict__ hs) {
  long long idx = (long long)blockIdx.x * 256 + threadIdx.x;
  if (idx >= (long long)kNS * 1280) return;
  int node = (int)(idx / 1280);
  int j = (int)(idx % 1280);
  float val;
  if (j < 128) {
    val = xs[(size_t)node * kD + j];
  } else if (j < 1152) {
    int which = (j - 128) >> 8;
    int c = (j - 128) & 255;
    const size_t seg = (size_t)kNS * 256;
    float cn = cnt[node];
    float m1 = 0.f, m2 = 0.f, m3 = 0.f, m4 = 0.f;
    if (cn > 0.f) {
      size_t o = (size_t)node * 256 + c;
      float inv = 1.f / cn;
      m1 = mom[o] * inv;
      m2 = mom[o + seg] * inv;
      m3 = mom[o + 2 * seg] * inv;
      m4 = mom[o + 3 * seg] * inv;
    }
    float var = m2 - m1 * m1;
    var = var >= 0.f ? var : 0.01f * var;  // F.leaky_relu default slope
    float stdv = sqrtf(var + 1e-6f);
    if (which == 0) val = m1;
    else if (which == 1) val = stdv;
    else if (which == 2) {
      float c3 = m3 - 3.f * m1 * m2 + 2.f * m1 * m1 * m1;
      val = (cn > 0.f) ? c3 / (stdv * stdv * stdv) : 0.f;
    } else {
      float c4 = m4 - 4.f * m1 * m3 + 6.f * m1 * m1 * m2 - 3.f * m1 * m1 * m1 * m1;
      val = (cn > 0.f) ? c4 / (stdv * stdv * stdv * stdv) : 0.f;
    }
  } else {
    val = xu[j - 1152];
  }
  hs[idx] = (bf16)val;
}

__global__ void ht_build_kernel(const float* __restrict__ xt, const float* __restrict__ agg,
                                const float* __restrict__ xu, bf16* __restrict__ ht) {
  long long idx = (long long)blockIdx.x * 256 + threadIdx.x;
  if (idx >= (long long)kNT * 512) return;
  int node = (int)(idx / 512);
  int j = (int)(idx % 512);
  float val;
  if (j < 128)      val = xt[(size_t)node * kD + j];
  else if (j < 384) val = agg[(size_t)node * 256 + (j - 128)];
  else              val = xu[j - 384];
  ht[idx] = (bf16)val;
}

// ---------------------------------------------------------------- global model
__global__ void colmean_kernel(const float* __restrict__ X, float* __restrict__ out, int rows) {
  int col = blockIdx.x;
  float s = 0.f;
  for (int r = threadIdx.x; r < rows; r += 256) s += X[(size_t)r * kD + col];
  __shared__ float red[256];
  red[threadIdx.x] = s;
  __syncthreads();
  for (int st = 128; st; st >>= 1) {
    if (threadIdx.x < st) red[threadIdx.x] += red[threadIdx.x + st];
    __syncthreads();
  }
  if (threadIdx.x == 0) out[col] = red[0] / (float)rows;
}

__global__ void global_kernel(const float* __restrict__ xu, const float* __restrict__ means,
                              const float* __restrict__ U1, const float* __restrict__ c1,
                              const float* __restrict__ U2, const float* __restrict__ c2,
                              const float* __restrict__ gg, float* __restrict__ xu_out) {
  __shared__ float hu[384], hid[384], dd[128], rr[1];
  int t = threadIdx.x;
  if (t < 384) hu[t] = (t < 128) ? xu[t] : means[t - 128];
  __syncthreads();
  if (t < 384) {
    float s = c1[t];
    for (int k = 0; k < 384; ++k) s += U1[(size_t)t * 384 + k] * hu[k];
    hid[t] = leaky(s, 0.2f);
  }
  __syncthreads();
  if (t < 128) {
    float s = c2[t];
    for (int k = 0; k < 384; ++k) s += U2[(size_t)t * 384 + k] * hid[k];
    dd[t] = s;
  }
  __syncthreads();
  if (t == 0) {
    float ss = 0.f;
    for (int i = 0; i < 128; ++i) ss += dd[i] * dd[i];
    rr[0] = rsqrtf(ss * (1.f / 128.f) + kEPS);
  }
  __syncthreads();
  if (t < 128) xu_out[t] = xu[t] + dd[t] * rr[0] * gg[t];
}

// ---------------------------------------------------------------- launch
extern "C" void kernel_launch(void* const* d_in, const int* in_sizes, int n_in,
                              void* d_out, int out_size, void* d_ws, size_t ws_size,
                              hipStream_t stream) {
  (void)in_sizes; (void)n_in; (void)out_size; (void)ws_size;
  const float* x_s = (const float*)d_in[0];
  const float* x_t = (const float*)d_in[1];
  const int*   ei  = (const int*)d_in[2];
  const float* ea  = (const float*)d_in[3];
  const float* x_u = (const float*)d_in[4];
  const float* e_Wsrc = (const float*)d_in[5];
  const float* e_Wtgt = (const float*)d_in[6];
  const float* e_Wedge = (const float*)d_in[7];
  const float* e_wattn = (const float*)d_in[8];
  const float* e_g = (const float*)d_in[9];
  const float* s_W1 = (const float*)d_in[10];
  const float* s_b1 = (const float*)d_in[11];
  const float* s_W2 = (const float*)d_in[12];
  const float* s_b2 = (const float*)d_in[13];
  const float* s_U1 = (const float*)d_in[14];
  const float* s_c1 = (const float*)d_in[15];
  const float* s_U2 = (const float*)d_in[16];
  const float* s_c2 = (const float*)d_in[17];
  const float* s_g  = (const float*)d_in[18];
  const float* t_W1 = (const float*)d_in[19];
  const float* t_b1 = (const float*)d_in[20];
  const float* t_W2 = (const float*)d_in[21];
  const float* t_b2 = (const float*)d_in[22];
  const float* t_U1 = (const float*)d_in[23];
  const float* t_c1 = (const float*)d_in[24];
  const float* t_U2 = (const float*)d_in[25];
  const float* t_c2 = (const float*)d_in[26];
  const float* t_g  = (const float*)d_in[27];
  const float* g_U1 = (const float*)d_in[28];
  const float* g_c1 = (const float*)d_in[29];
  const float* g_U2 = (const float*)d_in[30];
  const float* g_c2 = (const float*)d_in[31];
  const float* g_g  = (const float*)d_in[32];

  // output layout: xs | xt | edge_index | ea | xu
  float* xs_out = (float*)d_out;
  float* xt_out = xs_out + (size_t)kNS * kD;
  float* ei_out = xt_out + (size_t)kNT * kD;
  float* ea_out = ei_out + 2 * (size_t)kE;
  float* xu_out = ea_out + (size_t)kE * kD;

  // workspace layout
  char* ws = (char*)d_ws;
  size_t off = 0;
  auto alloc = [&](size_t bytes) -> void* {
    void* p = ws + off;
    off += (bytes + 255) & ~(size_t)255;
    return p;
  };
  bf16* WCAT = (bf16*)alloc((size_t)128 * 384 * 2);
  bf16* SW1  = (bf16*)alloc((size_t)256 * 256 * 2);
  bf16* SW2  = (bf16*)alloc((size_t)256 * 256 * 2);
  bf16* TW1  = (bf16*)alloc((size_t)256 * 256 * 2);
  bf16* TW2  = (bf16*)alloc((size_t)256 * 256 * 2);
  bf16* SU1  = (bf16*)alloc((size_t)1280 * 1280 * 2);
  bf16* SU2  = (bf16*)alloc((size_t)128 * 1280 * 2);
  bf16* TU1  = (bf16*)alloc((size_t)512 * 512 * 2);
  bf16* TU2  = (bf16*)alloc((size_t)128 * 512 * 2);
  float* ESC  = (float*)alloc((size_t)kE * 4);
  float* SMAX = (float*)alloc((size_t)kNS * 4);
  float* SSUM = (float*)alloc((size_t)kNS * 4);
  float* CNT  = (float*)alloc((size_t)kNS * 4);
  float* MEANS = (float*)alloc(256 * 4);
  float* REG_A = (float*)alloc((size_t)kE * kD * 4);        // h f32 -> hid_s/hid_t bf16
  float* REG_B = (float*)alloc((size_t)4 * kNS * 256 * 4);  // moments -> agg
  bf16*  REG_C = (bf16*)alloc((size_t)kNS * 1280 * 2);      // hs bf16 -> ht bf16
  float* DELTA = (float*)alloc((size_t)kNS * kD * 4);

  float* H   = REG_A;
  float* MOM = REG_B;
  float* AGG = REG_B;
  bf16* HS = REG_C;
  bf16* HT = REG_C;
  bf16* HID = (bf16*)REG_A;

  auto cdiv = [](long long a, long long b) { return (unsigned)((a + b - 1) / b); };

  // weights -> bf16
  wcat_kernel<<<cdiv(128 * 384, 256), 256, 0, stream>>>(e_Wsrc, e_Wtgt, e_Wedge, WCAT);
  f2bf_kernel<<<cdiv(256 * 256, 256), 256, 0, stream>>>(s_W1, SW1, 256 * 256);
  f2bf_kernel<<<cdiv(256 * 256, 256), 256, 0, stream>>>(s_W2, SW2, 256 * 256);
  f2bf_kernel<<<cdiv(256 * 256, 256), 256, 0, stream>>>(t_W1, TW1, 256 * 256);
  f2bf_kernel<<<cdiv(256 * 256, 256), 256, 0, stream>>>(t_W2, TW2, 256 * 256);
  f2bf_kernel<<<cdiv(1280 * 1280, 256), 256, 0, stream>>>(s_U1, SU1, 1280 * 1280);
  f2bf_kernel<<<cdiv(128 * 1280, 256), 256, 0, stream>>>(s_U2, SU2, 128 * 1280);
  f2bf_kernel<<<cdiv(512 * 512, 256), 256, 0, stream>>>(t_U1, TU1, 512 * 512);
  f2bf_kernel<<<cdiv(128 * 512, 256), 256, 0, stream>>>(t_U2, TU2, 128 * 512);

  // init accumulators
  hipMemsetAsync(SSUM, 0, (size_t)kNS * 4, stream);
  hipMemsetAsync(CNT, 0, (size_t)kNS * 4, stream);
  hipMemsetAsync(MOM, 0, (size_t)4 * kNS * 256 * 4, stream);
  fill_kernel<<<cdiv(kNS, 256), 256, 0, stream>>>(SMAX, kNS, -__builtin_inff());
  cnt_kernel<<<cdiv(kE, 256), 256, 0, stream>>>(ei, CNT);

  // ---- AttentionEdgeModel ----
  edge_h_kernel<<<kE / 64, 256, 0, stream>>>(x_s, x_t, ei, ea, WCAT, H);
  escore_kernel<<<cdiv(kE, 8), 256, 0, stream>>>(H, e_wattn, ESC);
  segmax_kernel<<<cdiv(kE, 256), 256, 0, stream>>>(ei, ESC, SMAX);
  expsum_kernel<<<cdiv(kE, 256), 256, 0, stream>>>(ei, ESC, SMAX, SSUM);
  alpha_kernel<<<cdiv(kE, 256), 256, 0, stream>>>(ei, ESC, SSUM);
  rmsres_kernel<<<cdiv(kE, 8), 256, 0, stream>>>(ea, H, ESC, e_g, ea_out, kE);

  // ---- SourceModel ----
  msg_kernel<<<kE / 64, 256, 0, stream>>>(x_t, ea_out, ei, SW1, s_b1, SW2, s_b2, MOM, 0);
  hs_build_kernel<<<cdiv((long long)kNS * 1280, 256), 256, 0, stream>>>(x_s, MOM, CNT, x_u, HS);
  gemm_kernel<bf16><<<dim3(cdiv(kNS, 64), 10), 256, 0, stream>>>(HS, SU1, s_c1, HID, kNS, 1280, 1280, 1);
  gemm_kernel<float><<<dim3(cdiv(kNS, 64), 1), 256, 0, stream>>>(HID, SU2, s_c2, DELTA, kNS, 128, 1280, 0);
  rmsres_kernel<<<cdiv(kNS, 8), 256, 0, stream>>>(x_s, DELTA, nullptr, s_g, xs_out, kNS);

  // ---- TargetModel ----
  hipMemsetAsync(AGG, 0, (size_t)kNT * 256 * 4, stream);
  msg_kernel<<<kE / 64, 256, 0, stream>>>(xs_out, ea_out, ei, TW1, t_b1, TW2, t_b2, AGG, 1);
  ht_build_kernel<<<cdiv((long long)kNT * 512, 256), 256, 0, stream>>>(x_t, AGG, x_u, HT);
  gemm_kernel<bf16><<<dim3(cdiv(kNT, 64), 4), 256, 0, stream>>>(HT, TU1, t_c1, HID, kNT, 512, 512, 1);
  gemm_kernel<float><<<dim3(cdiv(kNT, 64), 1), 256, 0, stream>>>(HID, TU2, t_c2, DELTA, kNT, 128, 512, 0);
  rmsres_kernel<<<cdiv(kNT, 8), 256, 0, stream>>>(x_t, DELTA, nullptr, t_g, xt_out, kNT);

  // ---- GlobalModel ----
  colmean_kernel<<<128, 256, 0, stream>>>(xs_out, MEANS, kNS);
  colmean_kernel<<<128, 256, 0, stream>>>(xt_out, MEANS + 128, kNT);
  global_kernel<<<1, 384, 0, stream>>>(x_u, MEANS, g_U1, g_c1, g_U2, g_c2, g_g, xu_out);

  // edge_index passthrough (value-cast to the f32 output buffer)
  i2f_kernel<<<cdiv(2 * kE, 256), 256, 0, stream>>>(ei, ei_out, 2 * kE);
}
